// DKOKernel_28587302322466
// MI455X (gfx1250) — matrix operand order repaired and checked by reading, not compiled, not run on hardware
//
#include <hip/hip_runtime.h>
#include <hip/hip_bf16.h>

typedef __attribute__((ext_vector_type(16))) _Float16 v16h;
typedef __attribute__((ext_vector_type(8)))  _Float16 v8h;
typedef __attribute__((ext_vector_type(8)))  float    v8f;

#define EPSV 1e-5f

// ---------------------------------------------------------------------------
// Kernel 0: pack an f32 weight matrix [N][K] into f16 B-operand slice images:
//   dst[ks][nt][lane][t] = w[nt*16 + (lane&15)][ks*32 + (lane>>4)*16 + t]
// One contiguous NT*1KB image per 32-wide k-step -> async-copyable verbatim.
// ---------------------------------------------------------------------------
__global__ __launch_bounds__(256) void k0_pack(const float* __restrict__ w,
                                               _Float16* __restrict__ dst,
                                               int Kdim, int NT, int total)
{
    for (int idx = blockIdx.x * 256 + threadIdx.x; idx < total;
         idx += gridDim.x * 256) {
        int t  = idx & 15;
        int l  = (idx >> 4) & 31;
        int nt = (idx >> 9) % NT;
        int ks = idx / (NT * 512);
        dst[idx] = (_Float16)w[(size_t)(nt * 16 + (l & 15)) * Kdim
                               + ks * 32 + ((l >> 4) << 4) + t];
    }
}

// ---------------------------------------------------------------------------
// Kernel 1: hx' = (x @ wx^T) * s1 ; hy' = (y @ wy^T) * s1 + c1   (stored f16)
//   s1 = g1 * rsqrt(v1+eps), c1 = (b1 - m1)*s1 + be1
// ---------------------------------------------------------------------------
__global__ __launch_bounds__(256) void k1_proj(
    const float* __restrict__ x, const float* __restrict__ y,
    const float* __restrict__ w1, const float* __restrict__ b1,
    const float* __restrict__ g1, const float* __restrict__ be1,
    const float* __restrict__ m1, const float* __restrict__ v1,
    _Float16* __restrict__ hxBuf, _Float16* __restrict__ hyBuf)
{
    int id  = blockIdx.x;
    bool isY = (id >= 1024);
    int row = id & 1023;                       // b*128 + i
    const float* src = (isY ? y : x) + (size_t)row * 128;

    __shared__ float sRow[128];
    int tid = threadIdx.x;
    if (tid < 128) sRow[tid] = src[tid];
    __syncthreads();

    _Float16* dst = (isY ? hyBuf : hxBuf) + (size_t)row * 512;
    int off = isY ? 128 : 0;
    for (int h = tid; h < 512; h += 256) {
        float s1 = g1[h] * rsqrtf(v1[h] + EPSV);
        const float* wr = w1 + (size_t)h * 256 + off;
        float acc = 0.f;
        #pragma unroll 8
        for (int f = 0; f < 128; ++f) acc = fmaf(sRow[f], wr[f], acc);
        float o = acc * s1;
        if (isY) o += (b1[h] - m1[h]) * s1 + be1[h];
        dst[h] = (_Float16)o;
    }
}

// ---------------------------------------------------------------------------
// Double-buffered async-staged WMMA GEMM.
//   sIn   : f16 activations, row-major [64][ldin] (M = 64 pixels, 4 M-tiles)
//   wPack : f16 pre-packed B slices, [KSTEPS][NT_TOTAL*512] halfs
//   sB0/1 : LDS double buffers for one slice each
// Wave owns M-tile `mtile`, N-tiles ntbase..ntbase+NT_WAVE-1.
// Weight slices stream via GLOBAL_LOAD_ASYNC_TO_LDS_B128 (ASYNCcnt),
// prefetching slice ks+1 while WMMA consumes slice ks.
// ---------------------------------------------------------------------------
template<int NT_TOTAL, int NT_WAVE, int KSTEPS>
__device__ inline void fused_gemm_async(const _Float16* sIn, int ldin,
                                        const _Float16* __restrict__ wPack,
                                        _Float16* sB0, _Float16* sB1,
                                        v8f* acc, int mtile, int ntbase,
                                        int tid, int lane)
{
    constexpr int SLICE = NT_TOTAL * 512;          // halfs per k-step image
    constexpr int NOPS  = SLICE / (8 * 256);       // 16B async ops per thread
    static_assert(NOPS >= 1, "slice must cover the block");

    #pragma unroll
    for (int n = 0; n < NT_WAVE; ++n)
        #pragma unroll
        for (int v = 0; v < 8; ++v) acc[n][v] = 0.f;

    __syncthreads();    // sB buffers free; sIn stores from all waves visible

    // prefetch slice 0 into sB0
    #pragma unroll
    for (int q = 0; q < NOPS; ++q) {
        int off = (tid + q * 256) * 8;
        unsigned ldsa = (unsigned)(uintptr_t)(sB0 + off);
        unsigned long long ga = (unsigned long long)(uintptr_t)(wPack + off);
        asm volatile("global_load_async_to_lds_b128 %0, %1, off"
                     :: "v"(ldsa), "v"(ga) : "memory");
    }

    for (int ks = 0; ks < KSTEPS; ++ks) {
        _Float16* cur = (ks & 1) ? sB1 : sB0;
        _Float16* nxt = (ks & 1) ? sB0 : sB1;

        if (ks + 1 < KSTEPS) {       // prefetch next slice, then wait for cur
            const _Float16* srcp = wPack + (size_t)(ks + 1) * SLICE;
            #pragma unroll
            for (int q = 0; q < NOPS; ++q) {
                int off = (tid + q * 256) * 8;
                unsigned ldsa = (unsigned)(uintptr_t)(nxt + off);
                unsigned long long ga = (unsigned long long)(uintptr_t)(srcp + off);
                asm volatile("global_load_async_to_lds_b128 %0, %1, off"
                             :: "v"(ldsa), "v"(ga) : "memory");
            }
            asm volatile("s_wait_asynccnt %0" :: "i"(NOPS) : "memory");
        } else {
            asm volatile("s_wait_asynccnt 0" ::: "memory");
        }
        __syncthreads();             // slice `cur` resident for all waves

        // A operand: two 16B LDS chunks per lane (16-bit 16x32 A layout)
        int r = lane & 15, hi = lane >> 4;
        const _Float16* ap = sIn + (size_t)(mtile * 16 + r) * ldin
                                 + ks * 32 + hi * 16;
        v8h a0 = *(const v8h*)ap;
        v8h a1 = *(const v8h*)(ap + 8);
        v16h A;
        #pragma unroll
        for (int t = 0; t < 8; ++t) { A[t] = a0[t]; A[t + 8] = a1[t]; }

        #pragma unroll
        for (int n = 0; n < NT_WAVE; ++n) {
            v16h Bv = *(const v16h*)(cur + (size_t)(ntbase + n) * 512 + lane * 16);
            acc[n] = __builtin_amdgcn_wmma_f32_16x16x32_f16(
                false, A, false, Bv, (short)0, acc[n], false, false);
        }
        __syncthreads();             // all waves done reading cur
    }
}

// C/D layout: VGPR v, lanes 0-15 -> M=v, lanes 16-31 -> M=v+8; N = lane&15.
template<int NT_WAVE>
__device__ inline void store_bn_relu(const v8f* acc, _Float16* sOut, int ldo,
                                     int mtile, int ntbase,
                                     const float* sScale, const float* sBias,
                                     int lane)
{
    int col = lane & 15, rb = (lane >> 4) * 8;
    #pragma unroll
    for (int n = 0; n < NT_WAVE; ++n) {
        int o = (ntbase + n) * 16 + col;
        float sc = sScale[o], bi = sBias[o];
        #pragma unroll
        for (int v = 0; v < 8; ++v) {
            float xv = fmaf(acc[n][v], sc, bi);
            sOut[(size_t)(mtile * 16 + rb + v) * ldo + o] = (_Float16)fmaxf(xv, 0.f);
        }
    }
}

// ---------------------------------------------------------------------------
// Kernel 2: fused MLP. grid = 256 blocks = (b, i-tile 16, j-quarter 32),
// block = 256 threads = 8 waves. Per j-chunk of 4: M=64 pixels (m=jj*16+ii).
// ---------------------------------------------------------------------------
__global__ __launch_bounds__(256) void k2_mlp(
    const _Float16* __restrict__ hxBuf, const _Float16* __restrict__ hyBuf,
    const _Float16* __restrict__ wp2, const _Float16* __restrict__ wp3,
    const _Float16* __restrict__ wp4,
    const float* __restrict__ b2, const float* __restrict__ g2,
    const float* __restrict__ be2, const float* __restrict__ m2,
    const float* __restrict__ v2,
    const float* __restrict__ b3, const float* __restrict__ g3,
    const float* __restrict__ be3, const float* __restrict__ m3,
    const float* __restrict__ v3,
    const float* __restrict__ b4,
    float* __restrict__ eBuf)
{
    __shared__ __align__(16) _Float16 bufA[64 * 520]; // h1 / later h3 (64x136)
    __shared__ __align__(16) _Float16 bufB[64 * 264]; // h2
    __shared__ __align__(16) _Float16 sB0[16 * 512];  // B slice buffer 0
    __shared__ __align__(16) _Float16 sB1[16 * 512];  // B slice buffer 1
    __shared__ __align__(16) _Float16 sHX[16 * 512];
    __shared__ __align__(16) _Float16 sHY[4 * 512];
    __shared__ float sS2[256], sC2[256], sS3[128], sC3[128], sB4[64];

    int tid = threadIdx.x, lane = tid & 31, wave = tid >> 5;
    int blk = blockIdx.x;
    int b = blk >> 5, rem = blk & 31, itile = rem >> 2, jq = rem & 3;
    int i0 = itile * 16, jbase = jq * 32;

    {   // fold BN constants
        int o = tid;
        if (o < 256) {
            float s = g2[o] * rsqrtf(v2[o] + EPSV);
            sS2[o] = s; sC2[o] = (b2[o] - m2[o]) * s + be2[o];
        }
        if (o < 128) {
            float s = g3[o] * rsqrtf(v3[o] + EPSV);
            sS3[o] = s; sC3[o] = (b3[o] - m3[o]) * s + be3[o];
        }
        if (o < 64) sB4[o] = b4[o];
    }
    {   // stage hx' tile: 16 rows x 512 halfs = 1024 uint4
        const uint4* src = (const uint4*)(hxBuf + ((size_t)b * 128 + i0) * 512);
        uint4* dst = (uint4*)sHX;
        for (int e = tid; e < 1024; e += 256) dst[e] = src[e];
    }

    int mt = wave & 3;      // M-tile (== jj within chunk)
    int ng = wave >> 2;     // N-group selector (0/1)

    for (int c = 0; c < 8; ++c) {
        int j0 = jbase + c * 4;
        __syncthreads();    // bufA / sHY free for reuse
        {   // stage hy' for 4 contiguous j rows: 256 uint4
            const uint4* src = (const uint4*)(hyBuf + ((size_t)b * 128 + j0) * 512);
            ((uint4*)sHY)[tid] = src[tid];
        }
        __syncthreads();
        // layer 1 (elementwise): h1 = relu(hx' + hy')
        for (int idx = tid; idx < 64 * 512; idx += 256) {
            int m = idx >> 9, k = idx & 511;
            float v = (float)sHX[((m & 15) << 9) + k] + (float)sHY[((m >> 4) << 9) + k];
            bufA[m * 520 + k] = (_Float16)fmaxf(v, 0.f);
        }
        // layer 2: 512 -> 256
        v8f acc2[8];
        fused_gemm_async<16, 8, 16>(bufA, 520, wp2, sB0, sB1, acc2, mt, ng * 8, tid, lane);
        store_bn_relu<8>(acc2, bufB, 264, mt, ng * 8, sS2, sC2, lane);
        // layer 3: 256 -> 128  (output aliases bufA; h1 is dead)
        v8f acc3[4];
        fused_gemm_async<8, 4, 8>(bufB, 264, wp3, sB0, sB1, acc3, mt, ng * 4, tid, lane);
        store_bn_relu<4>(acc3, bufA, 136, mt, ng * 4, sS3, sC3, lane);
        // layer 4: 128 -> 64
        v8f acc4[2];
        fused_gemm_async<4, 2, 4>(bufA, 136, wp4, sB0, sB1, acc4, mt, ng * 2, tid, lane);
        // epilogue: e = acc + b4 -> global
        {
            int col = lane & 15, rb = (lane >> 4) * 8;
            int j = j0 + mt;
            float* ebase = eBuf + (((size_t)(b * 8 + itile) * 128 + j) << 10);
            #pragma unroll
            for (int n = 0; n < 2; ++n) {
                int o = (ng * 2 + n) * 16 + col;
                float bb = sB4[o];
                #pragma unroll
                for (int v = 0; v < 8; ++v)
                    ebase[(size_t)(rb + v) * 64 + o] = acc4[n][v] + bb;
            }
        }
    }
}

// ---------------------------------------------------------------------------
// Kernel 3: deterministic reduction. block per (b,i), 128 threads.
//   s[k]   = sum_j e[b,i,j,k] ;  out[j] = sum_k e[b,i,j,k] * s[k]
// ---------------------------------------------------------------------------
__global__ __launch_bounds__(128) void k3_out(const float* __restrict__ eBuf,
                                              float* __restrict__ out)
{
    int b = blockIdx.x >> 7, i = blockIdx.x & 127;
    int itile = i >> 4, ii = i & 15;
    const float* base = eBuf + (((size_t)(b * 8 + itile) * 128) << 10) + ii * 64;

    __shared__ float sS[64];
    int tid = threadIdx.x;
    if (tid < 64) {
        float a = 0.f;
        for (int j = 0; j < 128; ++j) a += base[(size_t)j * 1024 + tid];
        sS[tid] = a;
    }
    __syncthreads();

    int j = tid;
    const float* ep = base + (size_t)j * 1024;
    float a = 0.f;
    #pragma unroll
    for (int k = 0; k < 64; ++k) a = fmaf(ep[k], sS[k], a);
    out[(size_t)blockIdx.x * 128 + j] = a;
}

// ---------------------------------------------------------------------------
extern "C" void kernel_launch(void* const* d_in, const int* in_sizes, int n_in,
                              void* d_out, int out_size, void* d_ws, size_t ws_size,
                              hipStream_t stream)
{
    (void)in_sizes; (void)n_in; (void)out_size; (void)ws_size;
    const float* x   = (const float*)d_in[0];
    const float* y   = (const float*)d_in[1];
    const float* w1  = (const float*)d_in[2];
    const float* b1  = (const float*)d_in[3];
    const float* g1  = (const float*)d_in[4];
    const float* be1 = (const float*)d_in[5];
    const float* m1  = (const float*)d_in[6];
    const float* v1  = (const float*)d_in[7];
    const float* w2  = (const float*)d_in[8];
    const float* b2  = (const float*)d_in[9];
    const float* g2  = (const float*)d_in[10];
    const float* be2 = (const float*)d_in[11];
    const float* m2  = (const float*)d_in[12];
    const float* v2  = (const float*)d_in[13];
    const float* w3  = (const float*)d_in[14];
    const float* b3  = (const float*)d_in[15];
    const float* g3  = (const float*)d_in[16];
    const float* be3 = (const float*)d_in[17];
    const float* m3  = (const float*)d_in[18];
    const float* v3  = (const float*)d_in[19];
    const float* w4  = (const float*)d_in[20];
    const float* b4  = (const float*)d_in[21];

    char* ws = (char*)d_ws;
    _Float16* hxBuf = (_Float16*)ws;                        // 1 MB
    _Float16* hyBuf = (_Float16*)(ws + (1u << 20));         // 1 MB
    float*    eBuf  = (float*)(ws + (2u << 20));            // 32 MB
    size_t woff = (size_t)34u << 20;
    _Float16* wp2 = (_Float16*)(ws + woff);                 // 256 KB (16 slices)
    _Float16* wp3 = (_Float16*)(ws + woff + 262144);        // 64 KB  (8 slices)
    _Float16* wp4 = (_Float16*)(ws + woff + 262144 + 65536);// 16 KB  (4 slices)
    float* out = (float*)d_out;

    k0_pack<<<128, 256, 0, stream>>>(w2, wp2, 512, 16, 131072);
    k0_pack<<<32, 256, 0, stream>>>(w3, wp3, 256, 8, 32768);
    k0_pack<<<8, 256, 0, stream>>>(w4, wp4, 128, 4, 8192);
    k1_proj<<<2048, 256, 0, stream>>>(x, y, w1, b1, g1, be1, m1, v1, hxBuf, hyBuf);
    k2_mlp<<<256, 256, 0, stream>>>(hxBuf, hyBuf, wp2, wp3, wp4,
                                    b2, g2, be2, m2, v2,
                                    b3, g3, be3, m3, v3,
                                    b4, eBuf);
    k3_out<<<1024, 128, 0, stream>>>(eBuf, out);
}
